// TrueMambaBlock_6493990551916
// MI455X (gfx1250) — matrix operand order repaired and checked by reading, not compiled, not run on hardware
//
#include <hip/hip_runtime.h>
#include <hip/hip_bf16.h>

// ---------------- CDNA5 WMMA types ----------------
typedef __attribute__((ext_vector_type(16))) __bf16 v16bf;
typedef __attribute__((ext_vector_type(8)))  __bf16 v8bf;
typedef __attribute__((ext_vector_type(8)))  float  v8f;

static __device__ __forceinline__ v8f wmma_bf16(v16bf a, v16bf b, v8f c) {
    // (neg_a, A, neg_b, B, c_mod, C, reuse_a, reuse_b)
    return __builtin_amdgcn_wmma_f32_16x16x32_bf16(false, a, false, b, (short)0, c, false, false);
}

// fp32 -> bf16 hi/lo split (round-to-nearest-even hi, residual lo).
struct bf2 { __bf16 hi, lo; };
static __device__ __forceinline__ bf2 bf_split(float x) {
    bf2 r;
    unsigned u  = __builtin_bit_cast(unsigned, x);
    unsigned hb = (u + 0x7FFFu + ((u >> 16) & 1u)) & 0xFFFF0000u;
    float fh    = __builtin_bit_cast(float, hb);
    r.hi        = __builtin_bit_cast(__bf16, (unsigned short)(hb >> 16));
    float fl    = x - fh;
    unsigned ul = __builtin_bit_cast(unsigned, fl);
    unsigned lb = (ul + 0x7FFFu + ((ul >> 16) & 1u));
    r.lo        = __builtin_bit_cast(__bf16, (unsigned short)(lb >> 16));
    return r;
}

// ---------------- activations ----------------
template<int ACT> static __device__ __forceinline__ float apply_act(float v) {
    if constexpr (ACT == 1)      return 0.5f * v * (1.0f + erff(v * 0.70710678118654752f)); // exact gelu
    else if constexpr (ACT == 2) return 1.0f / (1.0f + expf(-v));                            // sigmoid
    else if constexpr (ACT == 3) return v / (1.0f + expf(-v));                               // silu
    else                         return v;
}

// byte-offset loads: keep the GEP as base + zext(u32) so the backend
// emits GVS addressing (SGPR base + 32-bit VGPR offset), not v_add_nc_u64.
static __device__ __forceinline__ v8bf  ldb8 (const char* base, unsigned byteOff) {
    return *(const v8bf*)(base + byteOff);
}
static __device__ __forceinline__ v16bf ldb16(const char* base, unsigned byteOff) {
    return *(const v16bf*)(base + byteOff);
}

// ============================================================
// Pack x (f32, M x 256) -> hi/lo bf16, PADDED rows: each batch gets
// 2 leading zero rows (8 x 2050 rows) so the causal conv needs no guards.
// ============================================================
__global__ void pack_x(const float* __restrict__ x, __bf16* __restrict__ Xhi, __bf16* __restrict__ Xlo)
{
    const int rp  = blockIdx.x;           // padded row
    const int col = threadIdx.x;
    const int b   = rp / 2050;
    const int lp  = rp - b * 2050;
    const float v = (lp < 2) ? 0.0f : x[(unsigned)(b * 2048 + (lp - 2)) * 256u + col];
    const bf2 s = bf_split(v);
    const unsigned idx = (unsigned)rp * 256u + col;
    Xhi[idx] = s.hi; Xlo[idx] = s.lo;
}

// ============================================================
// Pack row-major W (K x N, f32) into fragment-major bf16:
// run = (c*N + n)*2 + h holds 16 contiguous bf16: W[c*32 + h*16 + e][n].
// ============================================================
__global__ void pack_w(const float* __restrict__ W, __bf16* __restrict__ Whp, __bf16* __restrict__ Wlp,
                       int N, int K)
{
    const int run   = blockIdx.x * blockDim.x + threadIdx.x;
    const int total = (K >> 5) * N * 2;
    if (run >= total) return;
    const int h  = run & 1;
    const int cn = run >> 1;
    const int n  = cn % N;
    const int c  = cn / N;
    const int k0 = c * 32 + h * 16;
    #pragma unroll
    for (int e = 0; e < 16; ++e) {
        const bf2 s = bf_split(W[(unsigned)(k0 + e) * (unsigned)N + n]);
        Whp[(unsigned)run * 16u + e] = s.hi;
        Wlp[(unsigned)run * 16u + e] = s.lo;
    }
}

// Same, but source is conv_W (o, i, t) viewed as W[k = t*256 + i][o]. K=768, N=256.
__global__ void pack_w_conv(const float* __restrict__ convW, __bf16* __restrict__ Whp, __bf16* __restrict__ Wlp)
{
    const int run   = blockIdx.x * blockDim.x + threadIdx.x;
    const int total = 24 * 256 * 2;
    if (run >= total) return;
    const int h  = run & 1;
    const int cn = run >> 1;
    const int n  = cn & 255;
    const int c  = cn >> 8;
    const int k0 = c * 32 + h * 16;
    #pragma unroll
    for (int e = 0; e < 16; ++e) {
        const int kg = k0 + e;
        const int t  = kg >> 8;
        const int i  = kg & 255;
        const bf2 s = bf_split(convW[(unsigned)n * 768u + i * 3 + t]);
        Whp[(unsigned)run * 16u + e] = s.hi;
        Wlp[(unsigned)run * 16u + e] = s.lo;
    }
}

// ============================================================
// GEMM: out[M,N] = act(A[M,K] @ W[K,N] + bias)
// Hot loop uses byte offsets off uniform bases -> GVS addressing, no
// 64-bit VGPR address math, no spills (launch_bounds relaxes VGPR cap).
// block = 128 (4 waves); block tile 16x256; wave tile 16x64;
// 3 bf16 WMMA per 32-deep chunk (hi*hi + hi*lo + lo*hi).
// ============================================================
template<int ACT, bool PAD_A, bool SPLIT_OUT>
__global__ void __launch_bounds__(128, 1)
gemm_bf16(const __bf16* __restrict__ Ahi, const __bf16* __restrict__ Alo,
          const __bf16* __restrict__ Bhp, const __bf16* __restrict__ Blp,
          const float* __restrict__ bias,
          float* __restrict__ outf, __bf16* __restrict__ Ohi, __bf16* __restrict__ Olo,
          int M, int N, int K)
{
    const int lane = threadIdx.x & 31;
    const int wave = threadIdx.x >> 5;
    const int row0 = blockIdx.x * 16;
    const int n0   = blockIdx.y * 256 + wave * 64;
    const int half = lane >> 4;
    const int m    = lane & 15;
    const int r    = row0 + m;
    const int rp   = PAD_A ? (r + 2 * (r >> 11) + 2) : r;   // padded-row mapping

    const char* AhiB = (const char*)Ahi;
    const char* AloB = (const char*)Alo;
    const char* BhpB = (const char*)Bhp;
    const char* BlpB = (const char*)Blp;

    // A: two 16B runs per fragment; byte offsets.
    unsigned arun0 = ((unsigned)rp * (unsigned)K + (unsigned)(half * 16)) * 2u;
    unsigned arun1 = arun0 + 16u;                 // +8 elements

    // B: packed fragment byte offsets; bump by cstride bytes per chunk.
    const unsigned cstride = (unsigned)N * 64u;   // N*32 elements * 2B
    unsigned boff[4];
    bool     cok[4];
    #pragma unroll
    for (int t = 0; t < 4; ++t) {
        const int col = n0 + t * 16 + m;
        cok[t]  = (col < N);
        boff[t] = ((unsigned)col * 2u + (unsigned)half) * 32u;
    }

    v8f z = {};
    v8f acc[4] = { z, z, z, z };

    const int nchunks = K >> 5;
    for (int c = 0; c < nchunks; ++c) {
        union { v16bf v; v8bf h[2]; } uh, ul;
        uh.h[0] = ldb8(AhiB, arun0);
        uh.h[1] = ldb8(AhiB, arun1);
        ul.h[0] = ldb8(AloB, arun0);
        ul.h[1] = ldb8(AloB, arun1);
        arun0 += 64u; arun1 += 64u;               // +32 elements
        const v16bf ahi = uh.v;
        const v16bf alo = ul.v;
        #pragma unroll
        for (int t = 0; t < 4; ++t) {
            v16bf bhi, blo;
            if (cok[t]) {
                bhi = ldb16(BhpB, boff[t]);
                blo = ldb16(BlpB, boff[t]);
            } else {
                v16bf bz = {};
                bhi = bz; blo = bz;
            }
            boff[t] += cstride;
            acc[t] = wmma_bf16(ahi, bhi, acc[t]);
            acc[t] = wmma_bf16(ahi, blo, acc[t]);
            acc[t] = wmma_bf16(alo, bhi, acc[t]);
        }
    }

    #pragma unroll
    for (int t = 0; t < 4; ++t) {
        const int col = n0 + t * 16 + m;
        if (col < N) {
            const float bv = bias[col];
            #pragma unroll
            for (int i = 0; i < 8; ++i) {
                const unsigned rr = (unsigned)(row0 + i + half * 8);
                const unsigned oidx = rr * (unsigned)N + (unsigned)col;
                const float v = apply_act<ACT>(acc[t][i] + bv);
                if constexpr (SPLIT_OUT) {
                    const bf2 s = bf_split(v);
                    Ohi[oidx] = s.hi;
                    Olo[oidx] = s.lo;
                } else {
                    outf[oidx] = v;
                }
            }
        }
    }
}

// ============================================================
// Causal conv as implicit GEMM over padded x: K = 3 taps x 256, N = 256.
// Pad rows make every tap read valid (zero) memory - no guards.
// ============================================================
__global__ void __launch_bounds__(128, 1)
conv_gemm_bf16(const __bf16* __restrict__ Xhi, const __bf16* __restrict__ Xlo,
               const __bf16* __restrict__ Whp, const __bf16* __restrict__ Wlp,
               const float* __restrict__ bias, float* __restrict__ out)
{
    const int lane = threadIdx.x & 31;
    const int wave = threadIdx.x >> 5;
    const int row0 = blockIdx.x * 16;
    const int n0   = wave * 64;
    const int half = lane >> 4;
    const int m    = lane & 15;
    const int r    = row0 + m;
    const int rb   = r + 2 * (r >> 11);   // padded row of (b, l-2)

    const char* XhiB = (const char*)Xhi;
    const char* XloB = (const char*)Xlo;
    const char* WhpB = (const char*)Whp;
    const char* WlpB = (const char*)Wlp;

    const unsigned cstride = 256u * 64u;
    unsigned boff[4];
    #pragma unroll
    for (int tt = 0; tt < 4; ++tt) {
        const int col = n0 + tt * 16 + m;
        boff[tt] = ((unsigned)col * 2u + (unsigned)half) * 32u;
    }

    v8f z = {};
    v8f acc[4] = { z, z, z, z };

    for (int t = 0; t < 3; ++t) {
        unsigned arun0 = ((unsigned)(rb + t) * 256u + (unsigned)(half * 16)) * 2u;
        unsigned arun1 = arun0 + 16u;
        #pragma unroll
        for (int c8 = 0; c8 < 8; ++c8) {
            union { v16bf v; v8bf h[2]; } uh, ul;
            uh.h[0] = ldb8(XhiB, arun0);
            uh.h[1] = ldb8(XhiB, arun1);
            ul.h[0] = ldb8(XloB, arun0);
            ul.h[1] = ldb8(XloB, arun1);
            arun0 += 64u; arun1 += 64u;
            const v16bf ahi = uh.v;
            const v16bf alo = ul.v;
            #pragma unroll
            for (int tt = 0; tt < 4; ++tt) {
                const v16bf bhi = ldb16(WhpB, boff[tt]);
                const v16bf blo = ldb16(WlpB, boff[tt]);
                boff[tt] += cstride;
                acc[tt] = wmma_bf16(ahi, bhi, acc[tt]);
                acc[tt] = wmma_bf16(ahi, blo, acc[tt]);
                acc[tt] = wmma_bf16(alo, bhi, acc[tt]);
            }
        }
    }

    #pragma unroll
    for (int tt = 0; tt < 4; ++tt) {
        const int col = n0 + tt * 16 + m;
        const float bv = bias[col];
        #pragma unroll
        for (int i = 0; i < 8; ++i) {
            const unsigned rr = (unsigned)(row0 + i + half * 8);
            out[rr * 256u + (unsigned)col] = acc[tt][i] + bv;
        }
    }
}

// ============================================================
// Per-row SSM prep: delta = softplus(delta_in . dt_W + dt_b);
// bar_A = exp(delta*A), disc = (bar_A-1)/(A+1e-6). One wave32 per row.
// ============================================================
__global__ void ssm_prep(const float* __restrict__ P, const float* __restrict__ dt_W,
                         const float* __restrict__ dt_b, const float* __restrict__ A_param,
                         float* __restrict__ barA, float* __restrict__ disc)
{
    const int lane = threadIdx.x & 31;
    const int row  = blockIdx.x * (blockDim.x >> 5) + (threadIdx.x >> 5);
    const float* pr = P + (size_t)row * 784;

    float s = 0.0f;
    #pragma unroll
    for (int i = lane; i < 256; i += 32) s += pr[i] * dt_W[i];
    #pragma unroll
    for (int off = 16; off > 0; off >>= 1) s += __shfl_xor(s, off, 32);

    const float zv    = s + dt_b[0];
    const float delta = (zv > 20.0f) ? zv : log1pf(expf(zv));

    if (lane < 16) {
        const float As = A_param[lane] * pr[768 + lane];
        const float bA = expf(delta * As);
        barA[(size_t)row * 16 + lane] = bA;
        disc[(size_t)row * 16 + lane] = (bA - 1.0f) / (As + 1e-6f);
    }
}

// ============================================================
// Selective scan: thread per (batch, d), h[16] in registers,
// bar_A/disc broadcast through LDS per step. 8 blocks x 256 threads.
// ============================================================
__global__ void ssm_scan(const float* __restrict__ x, const float* __restrict__ P,
                         const float* __restrict__ barA, const float* __restrict__ disc,
                         const float* __restrict__ Dp, float* __restrict__ ssm_out)
{
    const int b = blockIdx.x;
    const int d = threadIdx.x;
    float h[16];
    #pragma unroll
    for (int s = 0; s < 16; ++s) h[s] = 0.0f;
    const float Dd = Dp[d];

    __shared__ float sA[16];
    __shared__ float sD[16];

    for (int t = 0; t < 2048; ++t) {
        const size_t row = (size_t)b * 2048 + t;
        __syncthreads();
        if (d < 16) { sA[d] = barA[row * 16 + d]; sD[d] = disc[row * 16 + d]; }
        __syncthreads();
        const float Bt = P[row * 784 + 256 + d];
        const float Ct = P[row * 784 + 512 + d];
        const float xt = x[row * 256 + d];
        const float w  = Bt * xt;
        float hs = 0.0f;
        #pragma unroll
        for (int s = 0; s < 16; ++s) { h[s] = fmaf(sA[s], h[s], sD[s] * w); hs += h[s]; }
        ssm_out[row * 256 + d] = Ct * hs + Dd * xt;
    }
}

// ============================================================
// out = g1 * conv + g2 * ssm
// ============================================================
__global__ void combine_out(const float* __restrict__ gate, const float* __restrict__ conv,
                            const float* __restrict__ ssm, float* __restrict__ out)
{
    const unsigned i   = blockIdx.x * 256u + threadIdx.x;
    const unsigned row = i >> 8;
    const unsigned d   = i & 255u;
    const float g1 = gate[row * 512u + d];
    const float g2 = gate[row * 512u + 256u + d];
    out[i] = g1 * conv[i] + g2 * ssm[i];
}

// ============================================================
extern "C" void kernel_launch(void* const* d_in, const int* in_sizes, int n_in,
                              void* d_out, int out_size, void* d_ws, size_t ws_size,
                              hipStream_t stream)
{
    (void)in_sizes; (void)n_in; (void)out_size; (void)ws_size;
    const float* x       = (const float*)d_in[0];
    const float* proj_W  = (const float*)d_in[1];
    const float* proj_b  = (const float*)d_in[2];
    const float* A_param = (const float*)d_in[3];
    const float* D_param = (const float*)d_in[4];
    const float* dt_W    = (const float*)d_in[5];
    const float* dt_b    = (const float*)d_in[6];
    const float* conv_W  = (const float*)d_in[7];
    const float* conv_b  = (const float*)d_in[8];
    const float* gate_W1 = (const float*)d_in[9];
    const float* gate_b1 = (const float*)d_in[10];
    const float* gate_W2 = (const float*)d_in[11];
    const float* gate_b2 = (const float*)d_in[12];

    const int M  = 8 * 2048;     // 16384 logical rows
    const int MP = 8 * 2050;     // padded rows (2 zero rows per batch)

    char* w = (char*)d_ws;
    auto alloc = [&](size_t bytes) { char* p = w; w += (bytes + 255) & ~(size_t)255; return p; };

    __bf16* Xhi  = (__bf16*)alloc((size_t)MP * 256 * 2);
    __bf16* Xlo  = (__bf16*)alloc((size_t)MP * 256 * 2);
    __bf16* W1hp = (__bf16*)alloc((size_t)(256 / 32) * 1024 * 2 * 16 * 2);
    __bf16* W1lp = (__bf16*)alloc((size_t)(256 / 32) * 1024 * 2 * 16 * 2);
    __bf16* W2hp = (__bf16*)alloc((size_t)(1024 / 32) * 512 * 2 * 16 * 2);
    __bf16* W2lp = (__bf16*)alloc((size_t)(1024 / 32) * 512 * 2 * 16 * 2);
    __bf16* PWhp = (__bf16*)alloc((size_t)(256 / 32) * 784 * 2 * 16 * 2);
    __bf16* PWlp = (__bf16*)alloc((size_t)(256 / 32) * 784 * 2 * 16 * 2);
    __bf16* CWhp = (__bf16*)alloc((size_t)24 * 256 * 2 * 16 * 2);
    __bf16* CWlp = (__bf16*)alloc((size_t)24 * 256 * 2 * 16 * 2);
    char*   h1rg = alloc((size_t)M * 1024 * 2 * 2);          // H1 hi+lo (dead after gemm2)
    __bf16* H1hi = (__bf16*)h1rg;
    __bf16* H1lo = (__bf16*)(h1rg + (size_t)M * 1024 * 2);
    float*  P    = (float*)alloc((size_t)M * 784 * 4);
    float*  gate = (float*)alloc((size_t)M * 512 * 4);
    // overlay the dead H1 region after gemm2:
    float*  conv = (float*)h1rg;                              // M*256 f32
    float*  ssm  = conv + (size_t)M * 256;                    // M*256 f32
    float*  barA = ssm  + (size_t)M * 256;                    // M*16
    float*  disc = barA + (size_t)M * 16;                     // M*16

    const dim3 blk(128);

    // ---- one-time layout packs (hoists all bf16-split VALU out of the GEMMs) ----
    pack_x<<<dim3(MP), dim3(256), 0, stream>>>(x, Xhi, Xlo);
    pack_w<<<dim3((8 * 1024 * 2 + 255) / 256), dim3(256), 0, stream>>>(gate_W1, W1hp, W1lp, 1024, 256);
    pack_w<<<dim3((32 * 512 * 2 + 255) / 256), dim3(256), 0, stream>>>(gate_W2, W2hp, W2lp, 512, 1024);
    pack_w<<<dim3((8 * 784 * 2 + 255) / 256), dim3(256), 0, stream>>>(proj_W, PWhp, PWlp, 784, 256);
    pack_w_conv<<<dim3((24 * 256 * 2 + 255) / 256), dim3(256), 0, stream>>>(conv_W, CWhp, CWlp);

    // ---- gate MLP ----
    gemm_bf16<1, true,  true ><<<dim3(M / 16, 1024 / 256), blk, 0, stream>>>(
        Xhi, Xlo, W1hp, W1lp, gate_b1, nullptr, H1hi, H1lo, M, 1024, 256);
    gemm_bf16<2, false, false><<<dim3(M / 16,  512 / 256), blk, 0, stream>>>(
        H1hi, H1lo, W2hp, W2lp, gate_b2, gate, nullptr, nullptr, M, 512, 1024);

    // ---- SSM projection (N=784, last block column-guarded) ----
    gemm_bf16<3, true,  false><<<dim3(M / 16, (784 + 255) / 256), blk, 0, stream>>>(
        Xhi, Xlo, PWhp, PWlp, proj_b, P, nullptr, nullptr, M, 784, 256);

    // ---- causal conv (runs after gemm2, H1 region is free) ----
    conv_gemm_bf16<<<dim3(M / 16), blk, 0, stream>>>(Xhi, Xlo, CWhp, CWlp, conv_b, conv);

    // ---- delta / bar_A / disc, scan, combine ----
    ssm_prep<<<dim3(M / 8), dim3(256), 0, stream>>>(P, dt_W, dt_b, A_param, barA, disc);
    ssm_scan<<<dim3(8), dim3(256), 0, stream>>>(x, P, barA, disc, D_param, ssm);
    combine_out<<<dim3(M), dim3(256), 0, stream>>>(gate, conv, ssm, (float*)d_out);
}
